// WindowAttention_39092792328434
// MI455X (gfx1250) — matrix operand order repaired and checked by reading, not compiled, not run on hardware
//
#include <hip/hip_runtime.h>
#include <hip/hip_bf16.h>

typedef __bf16 bf16;
typedef __attribute__((ext_vector_type(16))) __bf16 v16bf;
typedef __attribute__((ext_vector_type(8)))  __bf16 v8bf;
typedef __attribute__((ext_vector_type(8)))  float  v8f;

#define DIMC   384
#define QKVN   1152
#define NHEADS 12
#define HD     32
#define NTOK   49
#define NPAD   64
#define BWIN   4096
#define NWIN   256

// ---------------- CDNA5 WMMA fragment helpers (wave32) ----------------
// A (16x32 bf16, MxK): lane l<16 -> row M=l, holds K=0..7 (elems 0..7) and
// K=16..23 (elems 8..15); lane l>=16 -> row M=l-16, K=8..15 and K=24..31.
__device__ __forceinline__ v16bf load_A(const bf16* __restrict__ base,
                                        int row0, int k, int ld) {
  const int lane = threadIdx.x & 31;
  const bf16* p = base + (row0 + (lane & 15)) * ld + k + ((lane >> 4) << 3);
  v8bf lo = *(const v8bf*)(p);        // K = kh..kh+7
  v8bf hi = *(const v8bf*)(p + 16);   // K = kh+16..kh+23
  return __builtin_shufflevector(lo, hi, 0,1,2,3,4,5,6,7,8,9,10,11,12,13,14,15);
}

// B (32x16 bf16, KxN), sourced from an N-major ("transposed") buffer where
// column n's K values are contiguous: lane l<16 -> col N=l, K=0..7 and K=16..23;
// lane l>=16 -> col N=l-16, K=8..15 and K=24..31. Two 16B loads per lane.
__device__ __forceinline__ v16bf load_Bt(const bf16* __restrict__ baseT,
                                         int col0, int k, int ldk) {
  const int lane = threadIdx.x & 31;
  const bf16* p = baseT + (col0 + (lane & 15)) * ldk + k + ((lane >> 4) << 3);
  v8bf lo = *(const v8bf*)(p);
  v8bf hi = *(const v8bf*)(p + 16);
  return __builtin_shufflevector(lo, hi, 0,1,2,3,4,5,6,7,8,9,10,11,12,13,14,15);
}

__device__ __forceinline__ v8f wmma_bf16(v16bf a, v16bf b, v8f c) {
  // D = A*B + C, f32 accumulate
  return __builtin_amdgcn_wmma_f32_16x16x32_bf16(false, a, false, b,
                                                 (short)0, c, false, false);
}

// ---------------- weight convert/transpose to bf16 ----------------
__global__ __launch_bounds__(256) void convert_weights(
    const float* __restrict__ qkv_w,  // [384][1152] row-major
    const float* __restrict__ proj_w, // [384][384]  row-major
    bf16* __restrict__ qkv_wt,        // [1152][384] (out-col major)
    bf16* __restrict__ proj_wt)       // [384][384]  (out-col major)
{
  const int i = blockIdx.x * 256 + threadIdx.x;
  if (i < DIMC * QKVN) {
    const int k = i / QKVN, n = i % QKVN;
    qkv_wt[n * DIMC + k] = (bf16)qkv_w[i];
  }
  const int j = i - DIMC * QKVN;
  if (j >= 0 && j < DIMC * DIMC) {
    const int k = j / DIMC, n = j % DIMC;
    proj_wt[n * DIMC + k] = (bf16)proj_w[j];
  }
}

// ---------------- fused window attention: 1 block == 1 window ----------------
__global__ __launch_bounds__(256) void win_attn_fused(
    const float* __restrict__ x,         // [4096,49,384]
    const float* __restrict__ mask,      // [256,49,49]
    const float* __restrict__ qkv_b,     // [1152]
    const float* __restrict__ proj_b,    // [384]
    const float* __restrict__ rel_table, // [169,12]
    const int*   __restrict__ rel_index, // [49,49]
    const bf16*  __restrict__ qkv_wt,    // [1152][384]
    const bf16*  __restrict__ proj_wt,   // [384][384]
    float* __restrict__ out)             // [4096,49,384]
{
  extern __shared__ bf16 smem[];
  bf16* sXO = smem;               // [64][384]  x tile, reused as attn output
  bf16* sQ  = smem + 1 * 24576;   // [64][384]  scaled q, row-major
  bf16* sK  = smem + 2 * 24576;   // [64][384]  k, row-major (== Kt for QK^T)
  bf16* sVt = smem + 3 * 24576;   // [384][64]  v transposed (col-of-V major)
  bf16* sP  = smem + 4 * 24576;   // [2][64][64] softmax probs (2 head slots)

  const int w      = blockIdx.x;
  const int tid    = threadIdx.x;
  const int lane   = tid & 31;
  const int wid    = tid >> 5;          // 8 waves
  const int lane15 = lane & 15;
  const int lhalf  = lane >> 4;
  const float scale = 0.17677669529663687f;  // 32^-0.5

  // ---- stage 0: x window -> LDS bf16, pad rows 49..63 with zero ----
  {
    const float4* xg = (const float4*)(x + (size_t)w * NTOK * DIMC);
    for (int i = tid; i < NTOK * DIMC / 4; i += 256) {
      float4 v = xg[i];
      const int b0 = i * 4;
      sXO[b0+0] = (bf16)v.x; sXO[b0+1] = (bf16)v.y;
      sXO[b0+2] = (bf16)v.z; sXO[b0+3] = (bf16)v.w;
    }
    for (int i = NTOK * DIMC + tid; i < NPAD * DIMC; i += 256) sXO[i] = (bf16)0.f;
  }
  __syncthreads();

  // ---- stage 1: qkv = x @ Wqkv + b ; scatter into sQ (scaled), sK, sVt ----
  {
    const int mrow0 = (wid & 3) * 16;       // 4 M-slabs of 16 rows
    const int jbeg  = (wid >> 2) * 36;      // 72 N-tiles split over 2 wave cols
    for (int jt = jbeg; jt < jbeg + 36; ++jt) {
      const int col0 = jt * 16;
      v8f acc = {};
      #pragma unroll
      for (int k = 0; k < DIMC; k += 32) {
        v16bf a = load_A(sXO, mrow0, k, DIMC);
        v16bf b = load_Bt(qkv_wt, col0, k, DIMC);
        acc = wmma_bf16(a, b, acc);
      }
      const int   n    = col0 + lane15;
      const float bias = qkv_b[n];
      const int   mb   = mrow0 + lhalf * 8;
      if (n < DIMC) {                       // Q (pre-scaled)
        #pragma unroll
        for (int r = 0; r < 8; ++r)
          sQ[(mb + r) * DIMC + n] = (bf16)((acc[r] + bias) * scale);
      } else if (n < 2 * DIMC) {            // K
        const int nn = n - DIMC;
        #pragma unroll
        for (int r = 0; r < 8; ++r)
          sK[(mb + r) * DIMC + nn] = (bf16)(acc[r] + bias);
      } else {                              // V, stored transposed [384][64]
        const int nn = n - 2 * DIMC;
        bf16* pv = sVt + nn * NPAD + mb;
        #pragma unroll
        for (int r = 0; r < 8; ++r) pv[r] = (bf16)(acc[r] + bias);
      }
    }
  }
  __syncthreads();

  // ---- stage 2: per-head attention; 2 heads in flight (4 waves each) ----
  const float* mwin = mask + (size_t)(w & (NWIN - 1)) * NTOK * NTOK;
  {
    const int row0  = (wid & 3) * 16;
    const int hslot = wid >> 2;
    bf16* sPh = sP + hslot * NPAD * NPAD;
    for (int it = 0; it < NHEADS / 2; ++it) {
      const int h  = it * 2 + hslot;
      const int hc = h * HD;

      // S = q_h @ k_h^T  (K = 32 -> single WMMA per 16x16 tile)
      v16bf aq = load_A(sQ, row0, hc, DIMC);
      v8f s[4];
      #pragma unroll
      for (int j = 0; j < 4; ++j) {
        v8f z = {};
        s[j] = wmma_bf16(aq, load_Bt(sK, j * 16, hc, DIMC), z);
      }

      // mask + clipped rel-bias + clip + row softmax (row lives in one lane-half)
      const int mb = row0 + lhalf * 8;
      #pragma unroll
      for (int r = 0; r < 8; ++r) {
        const int  mrow  = mb + r;
        const bool rowok = (mrow < NTOK);
        float vals[4];
        float mx = -3.0e30f;
        #pragma unroll
        for (int j = 0; j < 4; ++j) {
          const int nc = j * 16 + lane15;
          float v;
          if (rowok && nc < NTOK) {
            v = s[j][r] + mwin[mrow * NTOK + nc];
            float bb = rel_table[rel_index[mrow * NTOK + nc] * NHEADS + h];
            bb = fminf(fmaxf(bb, -5.f), 5.f);
            v = fminf(fmaxf(v + bb, -10.f), 10.f);
          } else {
            v = -3.0e30f;                   // padded cols -> prob 0
          }
          vals[j] = v;
          mx = fmaxf(mx, v);
        }
        #pragma unroll
        for (int d = 1; d < 16; d <<= 1) mx = fmaxf(mx, __shfl_xor(mx, d, 32));
        float sum = 0.f;
        #pragma unroll
        for (int j = 0; j < 4; ++j) { vals[j] = __expf(vals[j] - mx); sum += vals[j]; }
        #pragma unroll
        for (int d = 1; d < 16; d <<= 1) sum += __shfl_xor(sum, d, 32);
        const float inv = 1.f / sum;
        #pragma unroll
        for (int j = 0; j < 4; ++j)
          sPh[mrow * NPAD + j * 16 + lane15] = (bf16)(vals[j] * inv);
      }
      __syncthreads();

      // O_h = P @ v_h  (K = 64 -> two WMMA steps; N = 32 -> two tiles)
      v8f o0 = {}, o1 = {};
      #pragma unroll
      for (int kk = 0; kk < NPAD; kk += 32) {
        v16bf a = load_A(sPh, row0, kk, NPAD);
        o0 = wmma_bf16(a, load_Bt(sVt + hc * NPAD,  0, kk, NPAD), o0);
        o1 = wmma_bf16(a, load_Bt(sVt + hc * NPAD, 16, kk, NPAD), o1);
      }
      bf16* po = sXO + (row0 + lhalf * 8) * DIMC + hc + lane15;
      #pragma unroll
      for (int r = 0; r < 8; ++r) {
        po[r * DIMC]      = (bf16)o0[r];
        po[r * DIMC + 16] = (bf16)o1[r];
      }
      __syncthreads();   // protect sP slot reuse / final sXO handoff
    }
  }

  // ---- stage 3: out = attn_out @ Wproj + b (f32 straight to HBM) ----
  {
    const int mrow0 = (wid & 3) * 16;
    const int jbeg  = (wid >> 2) * 12;     // 24 N-tiles over 2 wave cols
    const int mb    = mrow0 + lhalf * 8;
    for (int jt = jbeg; jt < jbeg + 12; ++jt) {
      const int col0 = jt * 16;
      v8f acc = {};
      #pragma unroll
      for (int k = 0; k < DIMC; k += 32) {
        v16bf a = load_A(sXO, mrow0, k, DIMC);
        v16bf b = load_Bt(proj_wt, col0, k, DIMC);
        acc = wmma_bf16(a, b, acc);
      }
      const int   n  = col0 + lane15;
      const float pb = proj_b[n];
      float* po = out + (size_t)w * NTOK * DIMC + n;
      #pragma unroll
      for (int r = 0; r < 8; ++r) {
        const int mrow = mb + r;
        if (mrow < NTOK) po[mrow * DIMC] = acc[r] + pb;
      }
    }
  }
}

extern "C" void kernel_launch(void* const* d_in, const int* in_sizes, int n_in,
                              void* d_out, int out_size, void* d_ws, size_t ws_size,
                              hipStream_t stream) {
  (void)in_sizes; (void)n_in; (void)out_size; (void)ws_size;
  const float* x         = (const float*)d_in[0];
  const float* mask      = (const float*)d_in[1];
  const float* qkv_w     = (const float*)d_in[2];
  const float* qkv_b     = (const float*)d_in[3];
  const float* proj_w    = (const float*)d_in[4];
  const float* proj_b    = (const float*)d_in[5];
  const float* rel_table = (const float*)d_in[6];
  const int*   rel_index = (const int*)d_in[7];
  float* out = (float*)d_out;

  bf16* qkv_wt  = (bf16*)d_ws;
  bf16* proj_wt = (bf16*)((char*)d_ws + (size_t)QKVN * DIMC * sizeof(bf16));

  const int total = DIMC * QKVN + DIMC * DIMC;
  convert_weights<<<(total + 255) / 256, 256, 0, stream>>>(qkv_w, proj_w,
                                                           qkv_wt, proj_wt);

  const size_t lds_bytes = (size_t)(4 * 24576 + 2 * NPAD * NPAD) * sizeof(bf16); // 208 KB
  (void)hipFuncSetAttribute((const void*)win_attn_fused,
                            hipFuncAttributeMaxDynamicSharedMemorySize,
                            (int)lds_bytes);
  win_attn_fused<<<BWIN, 256, lds_bytes, stream>>>(
      x, mask, qkv_b, proj_b, rel_table, rel_index, qkv_wt, proj_wt, out);
}
